// SegmentationHead_7490422964528
// MI455X (gfx1250) — matrix operand order repaired and checked by reading, not compiled, not run on hardware
//
#include <hip/hip_runtime.h>
#include <hip/hip_bf16.h>

// ---------------------------------------------------------------------------
// SegmentationHead for MI455X (gfx1250, wave32, WMMA)
//
// Stage 1: 3x3 conv 256->8 via implicit GEMM, bf16 WMMA (f32 accum).
//          HBM floor: 268MB / 23.3TB/s ~ 11.5us; bf16 WMMA keeps compute
//          well under that, fp32 WMMA (K=4) would not by a comfortable margin.
// Stage 2: argmax one-hot + 8x8 mean pool -> U (unfold layout)
// Stage 3: nz row scales
// Stage 4: per-channel attention GEMM via fp32 WMMA 16x16x4
// Stage 5: fold + 1x1 conv + out = corr*y + y   (y recycled in d_out in place)
// ---------------------------------------------------------------------------

typedef __attribute__((ext_vector_type(16))) __bf16 v16bf;
typedef __attribute__((ext_vector_type(8)))  float  v8f;
typedef __attribute__((ext_vector_type(2)))  float  v2f;

#define HW   (512 * 512)     // 262144 pixels per channel
#define NPIX 262144

__device__ __forceinline__ unsigned short f2bf(float f) {
    union { float f; unsigned int u; } v; v.f = f;
    unsigned int r = v.u + 0x7FFFu + ((v.u >> 16) & 1u);   // round-nearest-even
    return (unsigned short)(r >> 16);
}

// ---------------------------------------------------------------------------
// Kernel 1: 3x3 conv (Cin=256 -> Cout=8, pad 1) + bias, bf16 WMMA.
// Workgroup = 512 threads (16 waves); tile = 16 rows x 16 cols of output.
// Each wave: one output row (16 pixels) x 16 channels (8 real, 8 zero-pad).
// K loop: 8 cin-blocks of 32; per block 9 taps -> 9 WMMAs -> 72 WMMAs/wave.
// ---------------------------------------------------------------------------
__global__ __launch_bounds__(512)
void conv3x3_wmma_kernel(const float* __restrict__ x,
                         const float* __restrict__ conv_w,
                         const float* __restrict__ conv_b,
                         float* __restrict__ y /* d_out first half */) {
    __shared__ unsigned short Xlds[18 * 18 * 32];   // [row][col][cin32], bf16
    __shared__ unsigned short Wlds[9 * 16 * 32];    // [tap][co16][cin32], bf16

    const int tid  = threadIdx.x;
    const int wid  = tid >> 5;          // wave id 0..15 -> output row in tile
    const int lane = tid & 31;

    const int th = blockIdx.x >> 5;     // 32 tiles in h
    const int tw = blockIdx.x & 31;     // 32 tiles in w
    const int h0 = th << 4;
    const int w0 = tw << 4;

    v8f acc = {};

    union Frag { v16bf v; unsigned int u[8]; };

    const int ln  = lane & 15;
    const int hiA = (lane >> 4) ? 8 : 0;    // A: K interleave 0-7/16-23 vs 8-15/24-31
    const int hiB = (lane >> 4) ? 16 : 0;   // unused; B loads use hiA's interleave like A

    for (int cb = 0; cb < 8; ++cb) {
        // ---- stage x halo tile (18x18) for this cin block, bf16 ----
        for (int idx = tid; idx < 18 * 18 * 32; idx += 512) {
            int c  = idx % 18;
            int r  = (idx / 18) % 18;
            int kk = idx / 324;
            int gh = h0 + r - 1;
            int gw = w0 + c - 1;
            float v = 0.0f;
            if ((unsigned)gh < 512u && (unsigned)gw < 512u)
                v = x[((cb * 32 + kk) * 512 + gh) * 512 + gw];
            Xlds[(r * 18 + c) * 32 + kk] = f2bf(v);
        }
        // ---- stage weights for this cin block, co padded 8->16 with zeros ----
        for (int idx = tid; idx < 9 * 16 * 32; idx += 512) {
            int ck  = idx & 31;
            int co  = (idx >> 5) & 15;
            int tap = idx >> 9;
            float v = (co < 8) ? conv_w[(co * 256 + cb * 32 + ck) * 9 + tap] : 0.0f;
            Wlds[(tap * 16 + co) * 32 + ck] = f2bf(v);
        }
        __syncthreads();

        // ---- 9 taps: A = x rows (pixels) x cin, B = cin x co ----
        for (int tap = 0; tap < 9; ++tap) {
            const int dh = tap / 3;
            const int dw = tap % 3;
            Frag a, b;
            const int rowbase = ((wid + dh) * 18 + (ln + dw)) * 32;
#pragma unroll
            for (int j = 0; j < 8; ++j) {
                int k = 2 * j + ((j >= 4) ? 8 : 0) + hiA;
                a.u[j] = *(const unsigned int*)&Xlds[rowbase + k];
            }
            const int wbase = (tap * 16 + ln) * 32;
#pragma unroll
            for (int j = 0; j < 8; ++j)
                b.u[j] = *(const unsigned int*)&Wlds[wbase + 2 * j + ((j >= 4) ? 8 : 0) + hiA];

            acc = __builtin_amdgcn_wmma_f32_16x16x32_bf16(
                false, a.v, false, b.v, (short)0, acc, false, false);
        }
        __syncthreads();
    }

    // ---- epilogue: D[m][n], m = 8*(lane/16)+r (pixel), n = lane%16 (co) ----
    if (ln < 8) {
        const float bias = conv_b[ln];
        const int hh = h0 + wid;
#pragma unroll
        for (int r = 0; r < 8; ++r) {
            int m = ((lane >> 4) << 3) + r;
            y[ln * NPIX + hh * 512 + (w0 + m)] = acc[r] + bias;
        }
    }
}

// ---------------------------------------------------------------------------
// Kernel 2: channel argmax one-hot + 8x8 mean pool, written in unfold order:
// U[c][p1*8+p2][hp*8+wp], U is [8][64][64].
// ---------------------------------------------------------------------------
__global__ __launch_bounds__(256)
void pool_argmax_kernel(const float* __restrict__ y, float* __restrict__ U) {
    int cell = blockIdx.x * 256 + threadIdx.x;     // 4096 pooled cells
    if (cell >= 4096) return;
    int i = cell >> 6;       // pooled row 0..63
    int j = cell & 63;       // pooled col 0..63
    int cnt[8] = {0, 0, 0, 0, 0, 0, 0, 0};
    for (int py = 0; py < 8; ++py) {
        for (int px = 0; px < 8; ++px) {
            int h = i * 8 + py, w = j * 8 + px;
            int off = h * 512 + w;
            float best = y[off];
            int bc = 0;
#pragma unroll
            for (int c = 1; c < 8; ++c) {
                float v = y[c * NPIX + off];
                if (v > best) { best = v; bc = c; }    // first-max like argmax
            }
            cnt[bc]++;
        }
    }
    int hp = i >> 3, p1 = i & 7, wp = j >> 3, p2 = j & 7;
#pragma unroll
    for (int c = 0; c < 8; ++c)
        U[(c * 64 + (p1 * 8 + p2)) * 64 + (hp * 8 + wp)] = (float)cnt[c] * (1.0f / 64.0f);
}

// ---------------------------------------------------------------------------
// Kernel 3: per (c,l1) row scale = 1 / (count_nonzero + 1e-5)
// ---------------------------------------------------------------------------
__global__ __launch_bounds__(256)
void nz_scale_kernel(const float* __restrict__ attn, float* __restrict__ scale) {
    int t = blockIdx.x * 256 + threadIdx.x;        // 32768 rows
    if (t >= 32768) return;
    const float* p = attn + (size_t)t * 64;
    int c = 0;
#pragma unroll 8
    for (int k = 0; k < 64; ++k) c += (p[k] != 0.0f) ? 1 : 0;
    scale[t] = 1.0f / ((float)c + 1e-5f);
}

// ---------------------------------------------------------------------------
// Kernel 4: att[c] = (attn[c]/nz) @ U[c]   (4096x64 @ 64x64, per channel)
// fp32 WMMA 16x16x4. Block = 128 thr (4 waves = 4 N-tiles of 16).
// 2048 blocks = 8 channels x 256 row tiles. 16 K-steps of 4.
// ---------------------------------------------------------------------------
__global__ __launch_bounds__(128)
void att_gemm_kernel(const float* __restrict__ attn,
                     const float* __restrict__ scale,
                     const float* __restrict__ U,
                     float* __restrict__ attw) {
    const int ch   = blockIdx.x >> 8;
    const int row0 = (blockIdx.x & 255) << 4;
    const int wid  = threadIdx.x >> 5;
    const int lane = threadIdx.x & 31;
    const int n0   = wid << 4;
    const int ln   = lane & 15;
    const int k0   = (lane >> 4) << 1;      // {0,2}

    const float* Arow = attn + ((size_t)(ch * 4096 + row0 + ln)) * 64;
    const float  sc   = scale[ch * 4096 + row0 + ln];
    const float* Bc   = U + ch * 64 * 64;

    v8f acc = {};
#pragma unroll
    for (int kb = 0; kb < 16; ++kb) {
        v2f a, b;
        a.x = Arow[kb * 4 + k0]     * sc;
        a.y = Arow[kb * 4 + k0 + 1] * sc;
        b.x = Bc[(kb * 4 + k0)     * 64 + n0 + ln];
        b.y = Bc[(kb * 4 + k0 + 1) * 64 + n0 + ln];
        acc = __builtin_amdgcn_wmma_f32_16x16x4_f32(
            false, a, false, b, (short)0, acc, false, false);
    }
#pragma unroll
    for (int r = 0; r < 8; ++r) {
        int m = ((lane >> 4) << 3) + r;
        attw[((size_t)(ch * 4096 + row0 + m)) * 64 + n0 + ln] = acc[r];
    }
}

// ---------------------------------------------------------------------------
// Kernel 5: fold + 1x1 conv + out = corr*y + y  (in place over y in d_out)
// ---------------------------------------------------------------------------
__global__ __launch_bounds__(256)
void fold_mix_kernel(const float* __restrict__ attw,
                     const float* __restrict__ corr_w,   // [8][8]
                     const float* __restrict__ corr_b,   // [8]
                     float* __restrict__ out /* d_out first half, holds y */) {
    int pix = blockIdx.x * 256 + threadIdx.x;            // 262144
    if (pix >= NPIX) return;
    int h = pix >> 9, w = pix & 511;
    int hq = h >> 3, p1 = h & 7, wq = w >> 3, p2 = w & 7;
    int l1 = hq * 64 + wq;
    int l2 = p1 * 8 + p2;

    float c0[8], yv[8];
#pragma unroll
    for (int c = 0; c < 8; ++c) {
        c0[c] = attw[((size_t)(c * 4096 + l1)) * 64 + l2];
        yv[c] = out[c * NPIX + pix];
    }
#pragma unroll
    for (int d = 0; d < 8; ++d) {
        float s = corr_b[d];
#pragma unroll
        for (int c = 0; c < 8; ++c) s += corr_w[d * 8 + c] * c0[c];
        out[d * NPIX + pix] = s * yv[d] + yv[d];
    }
}

// ---------------------------------------------------------------------------
extern "C" void kernel_launch(void* const* d_in, const int* in_sizes, int n_in,
                              void* d_out, int out_size, void* d_ws, size_t ws_size,
                              hipStream_t stream) {
    const float* x      = (const float*)d_in[0];   // [1,256,512,512]
    const float* attn   = (const float*)d_in[1];   // [1,8,4096,64]
    const float* conv_w = (const float*)d_in[2];   // [8,256,3,3]
    const float* conv_b = (const float*)d_in[3];   // [8]
    const float* corr_w = (const float*)d_in[4];   // [8,8,1,1]
    const float* corr_b = (const float*)d_in[5];   // [8]

    float* out = (float*)d_out;                    // [0..2M): out, [2M..4M): attentions

    // workspace layout (floats): attw 2,097,152 | U 32,768 | scale 32,768
    float* attw  = (float*)d_ws;
    float* U     = attw + 2097152;
    float* scale = U + 32768;

    // 1) conv3x3 + bias -> y in d_out first half (1024 tiles of 16x16)
    conv3x3_wmma_kernel<<<1024, 512, 0, stream>>>(x, conv_w, conv_b, out);
    // 2) argmax + pool -> U
    pool_argmax_kernel<<<16, 256, 0, stream>>>(out, U);
    // 3) nz scales
    nz_scale_kernel<<<128, 256, 0, stream>>>(attn, scale);
    // 4) attention GEMM -> attw
    att_gemm_kernel<<<2048, 128, 0, stream>>>(attn, scale, U, attw);
    // 5) fold + 1x1 + residual mix, in place over y
    fold_mix_kernel<<<1024, 256, 0, stream>>>(attw, corr_w, corr_b, out);
    // 6) second output of the tuple: attentions, unchanged
    hipMemcpyAsync(out + 2097152, attn, 2097152 * sizeof(float),
                   hipMemcpyDeviceToDevice, stream);
}
